// Block_6012954214590
// MI455X (gfx1250) — matrix operand order repaired and checked by reading, not compiled, not run on hardware
//
#include <hip/hip_runtime.h>
#include <hip/hip_bf16.h>
#include <math.h>

typedef __attribute__((ext_vector_type(16))) _Float16 v16h;
typedef __attribute__((ext_vector_type(8)))  float    v8f;

// ---------------- geometry constants ----------------
#define BATCH   8
#define HW      64            // H = W = 64 after pool
#define PHW     66            // padded spatial extent (1-px halo)
#define CCH     64            // channel count
#define MROWS   (BATCH*HW*HW) // 32768
#define DI      128           // mamba d_inner
#define LSEQ    (HW*HW)       // 4096
#define DSTATE  16
#define NCHUNK  32
#define CHLEN   (LSEQ/NCHUNK) // 128
#define PADSZ   (BATCH*PHW*PHW*CCH) // 2230272

// workspace layout (float offsets)
#define OFF_POOL 0L                         // 98304
#define OFF_Y1   98304L                     // 2097152 (y1 raw, later y4/final)
#define OFF_Y2   (OFF_Y1 + 2097152L)        // 2097152 (y2/act2, later x2)
#define OFF_Y3   (OFF_Y2 + 2097152L)        // 2097152 (act3 = x, residual)
#define OFF_XZ   (OFF_Y3 + 2097152L)        // 8388608
#define OFF_XSC  (OFF_XZ + 8388608L)        // 4194304 (xs_conv, then gated y)
#define OFF_DBL  (OFF_XSC + 4194304L)       // 1179648
#define OFF_STAT (OFF_DBL + 1179648L)       // 128
#define OFF_AUXA (OFF_STAT + 128L)          // 524288
#define OFF_AUXB (OFF_AUXA + 524288L)       // 524288
#define OFF_HIN  (OFF_AUXB + 524288L)       // 524288
#define OFF_PAD  (OFF_HIN + 524288L)        // 2230272 (padded act1)
#define OFF_WT   (OFF_PAD + 2230272L)       // 36864   (transposed 3x3 weights)

// ---------------- elementwise / small kernels ----------------

// maxpool 2x2: NCHW (8,3,128,128) -> NHWC (8,64,64,3)
__global__ void maxpool_kernel(const float* __restrict__ in, float* __restrict__ out) {
  int idx = blockIdx.x * blockDim.x + threadIdx.x;
  if (idx >= BATCH*HW*HW*3) return;
  int ci = idx % 3; int t = idx / 3;
  int x = t & 63; int y = (t >> 6) & 63; int b = t >> 12;
  long base = (((long)(b*3 + ci)*128) + y*2)*128 + x*2;
  float v = fmaxf(fmaxf(in[base], in[base+1]), fmaxf(in[base+128], in[base+129]));
  out[idx] = v;
}

// conv 7x7 pad 3, Cin=3 (NHWC) -> (8,64,64,64) NHWC, weights OIHW (64,3,7,7)
__global__ void conv7x7_kernel(const float* __restrict__ in, const float* __restrict__ w,
                               float* __restrict__ out) {
  int idx = blockIdx.x * blockDim.x + threadIdx.x;
  if (idx >= MROWS*CCH) return;
  int co = idx & 63; int x = (idx >> 6) & 63; int y = (idx >> 12) & 63; int b = idx >> 18;
  float s = 0.f;
  for (int ky = 0; ky < 7; ++ky) {
    int yy = y + ky - 3; if ((unsigned)yy >= 64u) continue;
    for (int kx = 0; kx < 7; ++kx) {
      int xx = x + kx - 3; if ((unsigned)xx >= 64u) continue;
      const float* p = in + (((long)(b*HW + yy)*HW + xx)*3);
      const float* wp = w + (long)co*3*49 + ky*7 + kx;
      s += p[0]*wp[0] + p[1]*wp[49] + p[2]*wp[98];
    }
  }
  out[idx] = s;
}

// per-channel mean/rstd over M rows of [M, 64]; one block per channel
__global__ void bnstats_kernel(const float* __restrict__ y, float* __restrict__ stats) {
  __shared__ float ssum[256], ssq[256];
  int c = blockIdx.x;
  float s = 0.f, q = 0.f;
  for (int i = threadIdx.x; i < MROWS; i += 256) {
    float v = y[(long)i*CCH + c];
    s += v; q += v*v;
  }
  ssum[threadIdx.x] = s; ssq[threadIdx.x] = q;
  __syncthreads();
  for (int off = 128; off > 0; off >>= 1) {
    if (threadIdx.x < off) { ssum[threadIdx.x] += ssum[threadIdx.x+off];
                             ssq[threadIdx.x]  += ssq[threadIdx.x+off]; }
    __syncthreads();
  }
  if (threadIdx.x == 0) {
    float mean = ssum[0] / (float)MROWS;
    float var  = ssq[0] / (float)MROWS - mean*mean;
    stats[2*c]   = mean;
    stats[2*c+1] = rsqrtf(var + 1e-5f);
  }
}

// y = relu(g*(y-mean)*rstd + b), in place, [M,64]
__global__ void bnapply_kernel(float* __restrict__ y, const float* __restrict__ stats,
                               const float* __restrict__ g, const float* __restrict__ b) {
  int idx = blockIdx.x * blockDim.x + threadIdx.x;
  if (idx >= MROWS*CCH) return;
  int c = idx & 63;
  float v = y[idx];
  v = g[c] * (v - stats[2*c]) * stats[2*c+1] + b[c];
  y[idx] = fmaxf(v, 0.f);
}

// BN + relu, writing into 1-px zero halo layout (8,66,66,64)
__global__ void bnapply_pad_kernel(const float* __restrict__ y, const float* __restrict__ stats,
                                   const float* __restrict__ g, const float* __restrict__ b,
                                   float* __restrict__ pad) {
  int idx = blockIdx.x * blockDim.x + threadIdx.x;
  if (idx >= MROWS*CCH) return;
  int c = idx & 63; int x = (idx >> 6) & 63; int yy = (idx >> 12) & 63; int bb = idx >> 18;
  float v = y[idx];
  v = g[c] * (v - stats[2*c]) * stats[2*c+1] + b[c];
  pad[(((long)(bb*PHW + yy + 1)*PHW) + x + 1)*CCH + c] = fmaxf(v, 0.f);
}

__global__ void zero_kernel(float* __restrict__ dst, int n) {
  int idx = blockIdx.x * blockDim.x + threadIdx.x;
  if (idx < n) dst[idx] = 0.f;
}

// fill n floats with *src (exact _prompt result: LN over 1 channel == lnb)
__global__ void fill_kernel(float* __restrict__ dst, const float* __restrict__ src, int n) {
  int idx = blockIdx.x * blockDim.x + threadIdx.x;
  if (idx < n) dst[idx] = src[0];
}

// OIHW (64,64,3,3) -> wT[n][k], k = (ky*3+kx)*64 + ci  (contiguous-k B matrix)
__global__ void repack_w2_kernel(const float* __restrict__ w2, float* __restrict__ wT) {
  int idx = blockIdx.x * blockDim.x + threadIdx.x;
  if (idx >= CCH*576) return;
  int k = idx % 576; int n = idx / 576;
  int ky = k / 192; int rem = k - ky*192; int kx = rem >> 6; int ci = rem & 63;
  wT[idx] = w2[((long)(n*CCH + ci)*3 + ky)*3 + kx];
}

// NHWC (8,64,64,64) -> NCHW flat output
__global__ void nhwc2nchw_kernel(const float* __restrict__ src, float* __restrict__ dst) {
  int idx = blockIdx.x * blockDim.x + threadIdx.x;
  if (idx >= MROWS*CCH) return;
  int x = idx & 63; int y = (idx >> 6) & 63; int c = (idx >> 12) & 63; int b = idx >> 18;
  dst[idx] = src[(((long)(b*HW + y)*HW + x) << 6) + c];
}

// depthwise causal conv (k=4) + bias + silu: reads xs half of xz [M,256] -> xsc [M,128]
__global__ void dwconv_silu_kernel(const float* __restrict__ xz, const float* __restrict__ cw,
                                   const float* __restrict__ cb, float* __restrict__ xsc) {
  int idx = blockIdx.x * blockDim.x + threadIdx.x;
  if (idx >= MROWS*DI) return;
  int d = idx & 127; int m = idx >> 7; int l = m & (LSEQ-1);
  float s = cb[d];
#pragma unroll
  for (int j = 0; j < 4; ++j) {
    int l2 = l - 3 + j;
    if (l2 >= 0) s += xz[(long)(m - 3 + j)*256 + d] * cw[d*4 + j];
  }
  xsc[idx] = s * (1.f / (1.f + __expf(-s)));  // silu
}

// ---------------- chunked selective scan (3 passes, exact) ----------------
__device__ __forceinline__ float softplusf(float t) {
  return (t > 20.f) ? t : log1pf(__expf(t));
}

// pass 1: per-chunk affine composition (aprod, bacc)
__global__ void scan1_kernel(const float* __restrict__ dbl, const float* __restrict__ xsc,
                             const float* __restrict__ dtw, const float* __restrict__ dtb,
                             const float* __restrict__ Alog,
                             float* __restrict__ auxA, float* __restrict__ auxB) {
  int b  = blockIdx.x >> 8;
  int dg = (blockIdx.x >> 5) & 7;
  int ch = blockIdx.x & 31;
  int dl = threadIdx.x >> 4;
  int s  = threadIdx.x & 15;
  int d  = dg*16 + dl;
  float Anf = -__expf(Alog[d*DSTATE + s]);
  float w0 = dtw[d*4+0], w1 = dtw[d*4+1], w2 = dtw[d*4+2], w3 = dtw[d*4+3];
  float tb = dtb[d];
  float ap = 1.f, bacc = 0.f;
  long mbase = (long)b*LSEQ + (long)ch*CHLEN;
  for (int l = 0; l < CHLEN; ++l) {
    long m = mbase + l;
    const float* q = dbl + m*36;
    float dt = softplusf(q[0]*w0 + q[1]*w1 + q[2]*w2 + q[3]*w3 + tb);
    float xv = xsc[m*DI + d];
    float a  = __expf(dt*Anf);
    float bb = dt*xv*q[4 + s];
    ap   = ap * a;
    bacc = a*bacc + bb;
  }
  long bds = (long)(b*DI + d)*DSTATE + s;
  auxA[bds*NCHUNK + ch] = ap;
  auxB[bds*NCHUNK + ch] = bacc;
}

// pass 2: exclusive scan over chunk summaries -> per-chunk initial h
__global__ void scan2_kernel(const float* __restrict__ auxA, const float* __restrict__ auxB,
                             float* __restrict__ hin) {
  int bds = blockIdx.x * blockDim.x + threadIdx.x;
  if (bds >= BATCH*DI*DSTATE) return;
  long base = (long)bds*NCHUNK;
  float h = 0.f;
  for (int c = 0; c < NCHUNK; ++c) {
    hin[base + c] = h;
    h = auxA[base + c]*h + auxB[base + c];
  }
}

// pass 3: re-run each chunk from exact h0, emit gated output y in place of xsc
__global__ void scan3_kernel(const float* __restrict__ dbl, float* __restrict__ xsc,
                             const float* __restrict__ xz, const float* __restrict__ dtw,
                             const float* __restrict__ dtb, const float* __restrict__ Alog,
                             const float* __restrict__ Dp, const float* __restrict__ hin) {
  int b  = blockIdx.x >> 8;
  int dg = (blockIdx.x >> 5) & 7;
  int ch = blockIdx.x & 31;
  int dl = threadIdx.x >> 4;
  int s  = threadIdx.x & 15;
  int d  = dg*16 + dl;
  float Anf = -__expf(Alog[d*DSTATE + s]);
  float w0 = dtw[d*4+0], w1 = dtw[d*4+1], w2 = dtw[d*4+2], w3 = dtw[d*4+3];
  float tb = dtb[d];
  float Dd = Dp[d];
  long bds = (long)(b*DI + d)*DSTATE + s;
  float h = hin[bds*NCHUNK + ch];
  long mbase = (long)b*LSEQ + (long)ch*CHLEN;
  for (int l = 0; l < CHLEN; ++l) {
    long m = mbase + l;
    const float* q = dbl + m*36;
    float dt = softplusf(q[0]*w0 + q[1]*w1 + q[2]*w2 + q[3]*w3 + tb);
    float xv = xsc[m*DI + d];
    float Bv = q[4 + s];
    float Cv = q[20 + s];
    h = __expf(dt*Anf)*h + dt*xv*Bv;
    float part = h * Cv;
    part += __shfl_xor(part, 1, 32);
    part += __shfl_xor(part, 2, 32);
    part += __shfl_xor(part, 4, 32);
    part += __shfl_xor(part, 8, 32);
    if (s == 0) {
      float zv = xz[m*256 + DI + d];
      float yv = (part + Dd*xv) * (zv * (1.f / (1.f + __expf(-zv))));
      xsc[m*DI + d] = yv;
    }
  }
}

// ---------------- WMMA GEMM kernels ----------------
__device__ __forceinline__ void pack8(v16h& f, int base, float4 a, float4 b) {
  f[base+0] = (_Float16)a.x; f[base+1] = (_Float16)a.y;
  f[base+2] = (_Float16)a.z; f[base+3] = (_Float16)a.w;
  f[base+4] = (_Float16)b.x; f[base+5] = (_Float16)b.y;
  f[base+6] = (_Float16)b.z; f[base+7] = (_Float16)b.w;
}

// D[M,N] = A[M,K] x W[N,K]^T (+res when HASRES). One wave -> 16x32 tile
// (A fragment reused, two accumulators). K%32==0, M%16==0. Out-of-range
// columns clamped on load (duplicate column, never stored).
template <bool HASRES>
__global__ void gemm_wmma_kernel(const float* __restrict__ A, const float* __restrict__ W,
                                 float* __restrict__ D, const float* __restrict__ res,
                                 int M, int N, int K) {
  int tilesN = (N + 15) >> 4;
  int pairsN = (tilesN + 1) >> 1;
  int wave = (blockIdx.x * blockDim.x + threadIdx.x) >> 5;
  if (wave >= (M >> 4) * pairsN) return;      // wave-uniform: EXEC stays all-ones
  int lane = threadIdx.x & 31;
  int tm = wave / pairsN, tp = wave % pairsN;
  int row   = tm*16 + (lane & 15);
  int hiA   = (lane & 16) ? 8  : 0;
  int hiB   = (lane & 16) ? 8  : 0;
  int ncol0 = tp*32 + (lane & 15);
  int ncol1 = ncol0 + 16;
  int nc0   = ncol0 < N ? ncol0 : N-1;        // clamp: safe duplicate column
  int nc1   = ncol1 < N ? ncol1 : N-1;
  const float* arow = A + (long)row*K;
  const float* b0r  = W + (long)nc0*K;
  const float* b1r  = W + (long)nc1*K;
  v8f acc0 = {}, acc1 = {};
  for (int kb = 0; kb < K; kb += 32) {
    __builtin_prefetch((const void*)(arow + kb + 128), 0, 3);
    const float4* ap = (const float4*)(arow + kb + hiA);
    const float4* b0 = (const float4*)(b0r  + kb + hiB);
    const float4* b1 = (const float4*)(b1r  + kb + hiB);
    v16h a, f0, f1;
    pack8(a, 0, ap[0], ap[1]);      // K = hiA + 0..7
    pack8(a, 8, ap[4], ap[5]);      // K = hiA + 16..23
    pack8(f0, 0, b0[0], b0[1]);     // K = hiB + 0..7
    pack8(f0, 8, b0[4], b0[5]);     // K = hiB + 16..23
    pack8(f1, 0, b1[0], b1[1]);
    pack8(f1, 8, b1[4], b1[5]);
    acc0 = __builtin_amdgcn_wmma_f32_16x16x32_f16(false, a, false, f0,
                                                  (short)0, acc0, false, false);
    acc1 = __builtin_amdgcn_wmma_f32_16x16x32_f16(false, a, false, f1,
                                                  (short)0, acc1, false, false);
  }
  int mbase = tm*16 + ((lane & 16) ? 8 : 0);
  if (ncol0 < N) {
#pragma unroll
    for (int i = 0; i < 8; ++i) {
      long o = (long)(mbase + i)*N + ncol0;
      float v = acc0[i];
      if (HASRES) v += res[o];
      D[o] = v;
    }
  }
  if (ncol1 < N) {
#pragma unroll
    for (int i = 0; i < 8; ++i) {
      long o = (long)(mbase + i)*N + ncol1;
      float v = acc1[i];
      if (HASRES) v += res[o];
      D[o] = v;
    }
  }
}

// conv 3x3 pad 1 as implicit GEMM on pre-padded NHWC act (8,66,66,64) with
// pre-transposed weights wT[n][k]. M=32768, K=576, N=64. One wave -> 16x32
// tile. All loads unconditional float4 (halo supplies zeros).
__global__ void conv3x3_wmma_kernel(const float* __restrict__ pad, const float* __restrict__ wT,
                                    float* __restrict__ out) {
  int wave = (blockIdx.x * blockDim.x + threadIdx.x) >> 5;
  int lane = threadIdx.x & 31;
  int tm = wave >> 1, tp = wave & 1;     // 2 column-pairs cover N=64
  int row = tm*16 + (lane & 15);
  int bb = row >> 12; int hw = row & 4095;
  int y0 = hw >> 6;   int x0 = hw & 63;
  int hiA   = (lane & 16) ? 8  : 0;
  int hiB   = (lane & 16) ? 8  : 0;
  int ncol0 = tp*32 + (lane & 15);
  int ncol1 = ncol0 + 16;
  const float* abase = pad + ((long)(bb*PHW + y0)*PHW + x0)*CCH; // (ky=0,kx=0) corner
  const float* b0r   = wT + (long)ncol0*576;
  const float* b1r   = wT + (long)ncol1*576;
  v8f acc0 = {}, acc1 = {};
  for (int kb = 0; kb < 576; kb += 32) {
    v16h a, f0, f1;
#pragma unroll
    for (int r = 0; r < 2; ++r) {                 // two 8-wide A runs, no bounds check
      int kkb = kb + hiA + r*16;
      int ky = kkb / 192; int rem = kkb - ky*192; int kx = rem >> 6; int ci = rem & 63;
      const float4* p = (const float4*)(abase + ((long)ky*PHW + kx)*CCH + ci);
      pack8(a, r*8, p[0], p[1]);
    }
    const float4* b0 = (const float4*)(b0r + kb + hiB);
    const float4* b1 = (const float4*)(b1r + kb + hiB);
    pack8(f0, 0, b0[0], b0[1]);
    pack8(f0, 8, b0[4], b0[5]);
    pack8(f1, 0, b1[0], b1[1]);
    pack8(f1, 8, b1[4], b1[5]);
    acc0 = __builtin_amdgcn_wmma_f32_16x16x32_f16(false, a, false, f0,
                                                  (short)0, acc0, false, false);
    acc1 = __builtin_amdgcn_wmma_f32_16x16x32_f16(false, a, false, f1,
                                                  (short)0, acc1, false, false);
  }
  int mbase = tm*16 + ((lane & 16) ? 8 : 0);
#pragma unroll
  for (int i = 0; i < 8; ++i) {
    out[(long)(mbase + i)*CCH + ncol0] = acc0[i];
    out[(long)(mbase + i)*CCH + ncol1] = acc1[i];
  }
}

// ---------------- host orchestration ----------------
static inline void run_gemm(const float* A, const float* W, float* D, const float* res,
                            int M, int N, int K, hipStream_t s) {
  int pairsN = (((N + 15) >> 4) + 1) >> 1;
  int waves = (M >> 4) * pairsN;
  int blocks = (waves * 32 + 255) / 256;
  if (res) gemm_wmma_kernel<true ><<<blocks, 256, 0, s>>>(A, W, D, res, M, N, K);
  else     gemm_wmma_kernel<false><<<blocks, 256, 0, s>>>(A, W, D, res, M, N, K);
}

extern "C" void kernel_launch(void* const* d_in, const int* in_sizes, int n_in,
                              void* d_out, int out_size, void* d_ws, size_t ws_size,
                              hipStream_t stream) {
  const float* x        = (const float*)d_in[0];
  const float* w1       = (const float*)d_in[1];
  const float* g1       = (const float*)d_in[2];
  const float* b1       = (const float*)d_in[3];
  const float* w2       = (const float*)d_in[4];
  const float* g2       = (const float*)d_in[5];
  const float* b2       = (const float*)d_in[6];
  const float* w3       = (const float*)d_in[7];
  const float* g3       = (const float*)d_in[8];
  const float* b3       = (const float*)d_in[9];
  const float* w4       = (const float*)d_in[10];
  const float* g4       = (const float*)d_in[11];
  const float* b4       = (const float*)d_in[12];
  const float* lnb1     = (const float*)d_in[15];
  const float* lnb2     = (const float*)d_in[18];
  const float* in_proj  = (const float*)d_in[19];
  const float* conv_w   = (const float*)d_in[20];
  const float* conv_b   = (const float*)d_in[21];
  const float* x_proj   = (const float*)d_in[22];
  const float* dt_w     = (const float*)d_in[23];
  const float* dt_b     = (const float*)d_in[24];
  const float* A_log    = (const float*)d_in[25];
  const float* Dp       = (const float*)d_in[26];
  const float* out_proj = (const float*)d_in[27];

  float* ws   = (float*)d_ws;
  float* pool = ws + OFF_POOL;
  float* y1   = ws + OFF_Y1;   // raw conv outputs, later y4/final NHWC
  float* y2   = ws + OFF_Y2;   // y2/act2, later x2
  float* y3   = ws + OFF_Y3;   // act3 = x (residual source)
  float* xz   = ws + OFF_XZ;
  float* xsc  = ws + OFF_XSC;
  float* dbl  = ws + OFF_DBL;
  float* stat = ws + OFF_STAT;
  float* auxA = ws + OFF_AUXA;
  float* auxB = ws + OFF_AUXB;
  float* hin  = ws + OFF_HIN;
  float* pad  = ws + OFF_PAD;
  float* wT   = ws + OFF_WT;

  float* out_x  = (float*)d_out;                 // 2097152 NCHW
  float* out_p1 = out_x + (long)MROWS*CCH;       // 32768
  float* out_p2 = out_p1 + MROWS;                // 32768

  const int EW = 256;
  // one-time prep (cheap, every call for determinism)
  zero_kernel<<<(PADSZ + EW-1)/EW, EW, 0, stream>>>(pad, PADSZ);
  repack_w2_kernel<<<(CCH*576 + EW-1)/EW, EW, 0, stream>>>(w2, wT);
  // stage 0: pool
  maxpool_kernel<<<(BATCH*HW*HW*3 + EW-1)/EW, EW, 0, stream>>>(x, pool);
  // stage 1: conv7x7 + BN + relu (into padded halo buffer)
  conv7x7_kernel<<<(MROWS*CCH + EW-1)/EW, EW, 0, stream>>>(pool, w1, y1);
  bnstats_kernel<<<CCH, EW, 0, stream>>>(y1, stat);
  bnapply_pad_kernel<<<(MROWS*CCH + EW-1)/EW, EW, 0, stream>>>(y1, stat, g1, b1, pad);
  // stage 2: conv3x3 (WMMA implicit GEMM, unconditional loads) + BN + relu
  conv3x3_wmma_kernel<<<(2048*2*32)/EW, EW, 0, stream>>>(pad, wT, y2);
  bnstats_kernel<<<CCH, EW, 0, stream>>>(y2, stat);
  bnapply_kernel<<<(MROWS*CCH + EW-1)/EW, EW, 0, stream>>>(y2, stat, g2, b2);
  // stage 3: conv1x1 (GEMM) + BN + relu  -> act3 (= x, NHWC == seq)
  run_gemm(y2, w3, y3, nullptr, MROWS, CCH, CCH, stream);
  bnstats_kernel<<<CCH, EW, 0, stream>>>(y3, stat);
  bnapply_kernel<<<(MROWS*CCH + EW-1)/EW, EW, 0, stream>>>(y3, stat, g3, b3);
  // p1: LN over a single channel is identically lnb1
  fill_kernel<<<(MROWS + EW-1)/EW, EW, 0, stream>>>(out_p1, lnb1, MROWS);
  // mamba
  run_gemm(y3, in_proj, xz, nullptr, MROWS, 2*DI, CCH, stream);            // xz = seq @ in_proj^T
  dwconv_silu_kernel<<<(MROWS*DI + EW-1)/EW, EW, 0, stream>>>(xz, conv_w, conv_b, xsc);
  run_gemm(xsc, x_proj, dbl, nullptr, MROWS, 36, DI, stream);              // dbl = xs @ x_proj^T
  // chunked selective scan (exact): 4096 serial steps -> 128 + 32 + 128
  scan1_kernel<<<BATCH*8*NCHUNK, 256, 0, stream>>>(dbl, xsc, dt_w, dt_b, A_log, auxA, auxB);
  scan2_kernel<<<(BATCH*DI*DSTATE + EW-1)/EW, EW, 0, stream>>>(auxA, auxB, hin);
  scan3_kernel<<<BATCH*8*NCHUNK, 256, 0, stream>>>(dbl, xsc, xz, dt_w, dt_b, A_log, Dp, hin);
  // out_proj with fused residual add (+act3) -> x2 in y2
  run_gemm(xsc, out_proj, y2, y3, MROWS, CCH, DI, stream);
  // p2: identically lnb2
  fill_kernel<<<(MROWS + EW-1)/EW, EW, 0, stream>>>(out_p2, lnb2, MROWS);
  // stage 4: conv1x1 + BN + relu
  run_gemm(y2, w4, y1, nullptr, MROWS, CCH, CCH, stream);
  bnstats_kernel<<<CCH, EW, 0, stream>>>(y1, stat);
  bnapply_kernel<<<(MROWS*CCH + EW-1)/EW, EW, 0, stream>>>(y1, stat, g4, b4);
  // final layout NHWC -> NCHW
  nhwc2nchw_kernel<<<(MROWS*CCH + EW-1)/EW, EW, 0, stream>>>(y1, out_x);
}
